// DynamicGATEncoder_9131100471791
// MI455X (gfx1250) — compile-verified, matrix operand order checked by reading, not run on hardware
//
#include <hip/hip_runtime.h>
#include <hip/hip_bf16.h>

#define N_NODES 50000
#define E0      800000
#define EPRIME  850000
#define CEILDIV(a,b) (((a)+(b)-1)/(b))

typedef __attribute__((ext_vector_type(16))) __bf16 v16bf;
typedef __attribute__((ext_vector_type(8)))  __bf16 v8bf;
typedef __attribute__((ext_vector_type(8)))  float  v8f;

// ---------- helpers ----------
__device__ __forceinline__ unsigned short f2bf(float f) {
  unsigned b = __float_as_uint(f);
  b += 0x7FFFu + ((b >> 16) & 1u);          // round-to-nearest-even
  return (unsigned short)(b >> 16);
}
// order-preserving float<->uint map for atomicMax on floats
__device__ __forceinline__ unsigned fkey(float f) {
  unsigned b = __float_as_uint(f);
  return (b & 0x80000000u) ? ~b : (b | 0x80000000u);
}
__device__ __forceinline__ float fkey_inv(unsigned k) {
  unsigned b = (k & 0x80000000u) ? (k & 0x7FFFFFFFu) : ~k;
  return __uint_as_float(b);
}

// ---------- eattr mean (sum; divided by E0 at use site) ----------
__global__ __launch_bounds__(256) void sum_kernel(const float* __restrict__ v,
                                                  float* __restrict__ out, int n) {
  __shared__ float sh[256];
  float s = 0.f;
  for (int i = blockIdx.x * blockDim.x + threadIdx.x; i < n; i += gridDim.x * blockDim.x)
    s += v[i];
  sh[threadIdx.x] = s;
  __syncthreads();
  for (int o = 128; o > 0; o >>= 1) {
    if (threadIdx.x < (unsigned)o) sh[threadIdx.x] += sh[threadIdx.x + o];
    __syncthreads();
  }
  if (threadIdx.x == 0) atomicAdd(out, sh[0]);
}

// ---------- conversions for WMMA operands ----------
__global__ __launch_bounds__(256) void cvt_f32_bf16_kernel(const float* __restrict__ src,
                                                           unsigned short* __restrict__ dst, int n) {
  int i = blockIdx.x * blockDim.x + threadIdx.x;
  if (i < n) dst[i] = f2bf(src[i]);
}
// W [K,Nc] f32 -> Wt [Nc,K] bf16 (so B-operand loads are contiguous in K)
__global__ __launch_bounds__(256) void cvt_transpose_w_kernel(const float* __restrict__ W,
                                                              unsigned short* __restrict__ Wt,
                                                              int K, int Nc) {
  int i = blockIdx.x * blockDim.x + threadIdx.x;
  if (i >= K * Nc) return;
  int k = i / Nc, n = i - k * Nc;
  Wt[(size_t)n * K + k] = f2bf(W[i]);
}

// ---------- WMMA GEMM: C[M,Nc] = A[M,K](bf16) * Bt[Nc,K]^T(bf16) + bias ----------
__global__ __launch_bounds__(256) void wmma_gemm_bias_kernel(
    const __bf16* __restrict__ A, const __bf16* __restrict__ Bt,
    const float* __restrict__ bias, float* __restrict__ C,
    int Mtiles, int Ntiles, int K, int Nc) {
  int wave = threadIdx.x >> 5;
  int lane = threadIdx.x & 31;
  int tile = blockIdx.x * 8 + wave;             // 8 waves / 256-thread block
  if (tile >= Mtiles * Ntiles) return;          // wave-uniform: EXEC stays all-ones
  int tm = tile / Ntiles, tn = tile - tm * Ntiles;
  int rc = lane & 15;                           // A-row / B-col within tile
  int kb = (lane >> 4) * 8;                     // K sub-block per lane half
  const __bf16* arow = A  + (size_t)(tm * 16 + rc) * K;
  const __bf16* brow = Bt + (size_t)(tn * 16 + rc) * K;
  v8f acc = {};
  for (int k0 = 0; k0 < K; k0 += 32) {
    v8bf alo = *(const v8bf*)(arow + k0 + kb);
    v8bf ahi = *(const v8bf*)(arow + k0 + kb + 16);
    v8bf blo = *(const v8bf*)(brow + k0 + kb);
    v8bf bhi = *(const v8bf*)(brow + k0 + kb + 16);
    v16bf a, b;
#pragma unroll
    for (int i = 0; i < 8; ++i) { a[i] = alo[i]; a[8 + i] = ahi[i];
                                  b[i] = blo[i]; b[8 + i] = bhi[i]; }
    acc = __builtin_amdgcn_wmma_f32_16x16x32_bf16(false, a, false, b,
                                                  (short)0, acc, false, false);
  }
  int colg  = tn * 16 + rc;
  float bv  = bias[colg];
  int rbase = tm * 16 + ((lane >> 4) * 8);
#pragma unroll
  for (int vg = 0; vg < 8; ++vg)
    C[(size_t)(rbase + vg) * Nc + colg] = acc[vg] + bv;
}

// ---------- pass 1: edge attention scores + segment max ----------
__global__ __launch_bounds__(256) void edge_score_kernel(
    const int* __restrict__ src, const int* __restrict__ dst,
    const float* __restrict__ eattr, const float* __restrict__ sum_ea,
    const float* __restrict__ xl, const float* __restrict__ xr,
    const float* __restrict__ We, const float* __restrict__ att,
    float* __restrict__ ebuf, unsigned* __restrict__ emax, int H, int hc) {
  int e = blockIdx.x * blockDim.x + threadIdx.x;
  if (e >= EPRIME) return;
  int s, d; float ea;
  if (e < E0) { s = src[e]; d = dst[e]; ea = eattr[e]; }
  else        { s = e - E0; d = s;      ea = sum_ea[0] * (1.0f / (float)E0); }
  const float4* xl4 = (const float4*)(xl + (size_t)s * hc);
  const float4* xr4 = (const float4*)(xr + (size_t)d * hc);
  const float4* we4 = (const float4*)We;
  const float4* at4 = (const float4*)att;
  for (int h = 0; h < H; ++h) {
    float sc = 0.f;
#pragma unroll
    for (int j4 = 0; j4 < 8; ++j4) {            // C = 32 -> 8 float4
      int j = h * 8 + j4;
      float4 a = xl4[j], b = xr4[j], w = we4[j], t = at4[j];
      float m;
      m = a.x + b.x + ea * w.x; m = m > 0.f ? m : 0.2f * m; sc += m * t.x;
      m = a.y + b.y + ea * w.y; m = m > 0.f ? m : 0.2f * m; sc += m * t.y;
      m = a.z + b.z + ea * w.z; m = m > 0.f ? m : 0.2f * m; sc += m * t.z;
      m = a.w + b.w + ea * w.w; m = m > 0.f ? m : 0.2f * m; sc += m * t.w;
    }
    ebuf[(size_t)e * H + h] = sc;
    atomicMax(&emax[(size_t)d * H + h], fkey(sc));
  }
}

// ---------- pass 2: exp(e - max) + segment sum ----------
__global__ __launch_bounds__(256) void edge_exp_kernel(
    const int* __restrict__ dst, float* __restrict__ ebuf,
    const unsigned* __restrict__ emax, float* __restrict__ den, int H) {
  int t = blockIdx.x * blockDim.x + threadIdx.x;
  if (t >= EPRIME * H) return;
  int e = t / H, h = t - e * H;
  int d = (e < E0) ? dst[e] : (e - E0);
  float mx = fkey_inv(emax[(size_t)d * H + h]);
  float ex = __expf(ebuf[t] - mx);
  ebuf[t] = ex;
  atomicAdd(&den[(size_t)d * H + h], ex);
}

// ---------- pass 3: alpha-weighted scatter-aggregate ----------
__global__ __launch_bounds__(256) void edge_aggr_kernel(
    const int* __restrict__ src, const int* __restrict__ dst,
    const float* __restrict__ ebuf, const float* __restrict__ den,
    const float* __restrict__ xl, float* __restrict__ out, int H, int hc) {
  int e = blockIdx.x * blockDim.x + threadIdx.x;
  if (e >= EPRIME) return;
  int s, d;
  if (e < E0) { s = src[e]; d = dst[e]; } else { s = e - E0; d = s; }
  for (int h = 0; h < H; ++h) {
    float alpha = ebuf[(size_t)e * H + h] / (den[(size_t)d * H + h] + 1e-16f);
    const float4* xs = (const float4*)(xl + (size_t)s * hc) + h * 8;
    float* od = out + (size_t)d * hc + h * 32;
#pragma unroll
    for (int c4 = 0; c4 < 8; ++c4) {
      float4 v = xs[c4];
      atomicAdd(&od[c4 * 4 + 0], alpha * v.x);
      atomicAdd(&od[c4 * 4 + 1], alpha * v.y);
      atomicAdd(&od[c4 * 4 + 2], alpha * v.z);
      atomicAdd(&od[c4 * 4 + 3], alpha * v.w);
    }
  }
}

// ---------- bias + ELU ----------
__global__ __launch_bounds__(256) void bias_elu_kernel(float* __restrict__ y,
                                                       const float* __restrict__ bias,
                                                       int n, int hc) {
  int i = blockIdx.x * blockDim.x + threadIdx.x;
  if (i >= n) return;
  float v = y[i] + bias[i % hc];                 // hc is a power of two
  y[i] = v > 0.f ? v : (__expf(v) - 1.f);
}

extern "C" void kernel_launch(void* const* d_in, const int* in_sizes, int n_in,
                              void* d_out, int out_size, void* d_ws, size_t ws_size,
                              hipStream_t stream) {
  (void)in_sizes; (void)n_in; (void)out_size; (void)ws_size;
  const float* x     = (const float*)d_in[0];
  const int*   ei    = (const int*)d_in[1];
  const float* eattr = (const float*)d_in[2];
  const int* src = ei;
  const int* dst = ei + E0;

  // ---- workspace carve (256B aligned) ----
  char* ws = (char*)d_ws;
  size_t off = 0;
  auto carve = [&](size_t bytes) -> void* {
    void* p = ws + off;
    off = (off + bytes + 255) & ~(size_t)255;
    return p;
  };
  unsigned short* a_bf = (unsigned short*)carve((size_t)N_NODES * 128 * 2);
  unsigned short* wlT  = (unsigned short*)carve(128 * 128 * 2);
  unsigned short* wrT  = (unsigned short*)carve(128 * 128 * 2);
  float* XL     = (float*)carve((size_t)N_NODES * 128 * 4);
  float* XR     = (float*)carve((size_t)N_NODES * 128 * 4);
  float* HA     = (float*)carve((size_t)N_NODES * 128 * 4);
  float* HB     = (float*)carve((size_t)N_NODES * 128 * 4);
  float* ebuf   = (float*)carve((size_t)EPRIME * 4 * 4);
  unsigned* emax = (unsigned*)carve((size_t)N_NODES * 4 * 4);
  float* den    = (float*)carve((size_t)N_NODES * 4 * 4);
  float* sum_ea = (float*)carve(256);

  // mean of edge_attr (as sum; scaled at use)
  hipMemsetAsync(sum_ea, 0, sizeof(float), stream);
  sum_kernel<<<256, 256, 0, stream>>>(eattr, sum_ea, E0);

  static const int spec_in[6] = {128, 128, 32, 128, 32, 128};
  static const int spec_H [6] = {4, 1, 4, 1, 4, 1};

  const float* hin = x;
  for (int L = 0; L < 6; ++L) {
    int in = spec_in[L];
    int H  = spec_H[L];
    int hc = H * 32;
    int base = 3 + 7 * L;   // per-conv params: Wl, bl, Wr, br, We, att, bias
    const float* Wl   = (const float*)d_in[base + 0];
    const float* bl   = (const float*)d_in[base + 1];
    const float* Wr   = (const float*)d_in[base + 2];
    const float* br   = (const float*)d_in[base + 3];
    const float* We   = (const float*)d_in[base + 4];
    const float* att  = (const float*)d_in[base + 5];
    const float* bias = (const float*)d_in[base + 6];

    float* obuf = (L == 5) ? (float*)d_out : ((L & 1) ? HB : HA);

    // 1) convert activations + weights to bf16 (weights transposed)
    int nIn = N_NODES * in;
    cvt_f32_bf16_kernel<<<CEILDIV(nIn, 256), 256, 0, stream>>>(hin, a_bf, nIn);
    cvt_transpose_w_kernel<<<CEILDIV(in * hc, 256), 256, 0, stream>>>(Wl, wlT, in, hc);
    cvt_transpose_w_kernel<<<CEILDIV(in * hc, 256), 256, 0, stream>>>(Wr, wrT, in, hc);

    // 2) WMMA GEMMs: XL = h@Wl+bl, XR = h@Wr+br
    int Mtiles = N_NODES / 16, Ntiles = hc / 16;
    int tiles = Mtiles * Ntiles;
    wmma_gemm_bias_kernel<<<CEILDIV(tiles, 8), 256, 0, stream>>>(
        (const __bf16*)a_bf, (const __bf16*)wlT, bl, XL, Mtiles, Ntiles, in, hc);
    wmma_gemm_bias_kernel<<<CEILDIV(tiles, 8), 256, 0, stream>>>(
        (const __bf16*)a_bf, (const __bf16*)wrT, br, XR, Mtiles, Ntiles, in, hc);

    // 3) segment softmax + aggregate (L2-resident working set on MI455X)
    hipMemsetAsync(emax, 0, (size_t)N_NODES * H * 4, stream);
    hipMemsetAsync(den,  0, (size_t)N_NODES * H * 4, stream);
    hipMemsetAsync(obuf, 0, (size_t)N_NODES * hc * 4, stream);
    edge_score_kernel<<<CEILDIV(EPRIME, 256), 256, 0, stream>>>(
        src, dst, eattr, sum_ea, XL, XR, We, att, ebuf, emax, H, hc);
    edge_exp_kernel<<<CEILDIV(EPRIME * H, 256), 256, 0, stream>>>(dst, ebuf, emax, den, H);
    edge_aggr_kernel<<<CEILDIV(EPRIME, 256), 256, 0, stream>>>(
        src, dst, ebuf, den, XL, obuf, H, hc);

    // 4) bias + ELU (in place)
    bias_elu_kernel<<<CEILDIV(N_NODES * hc, 256), 256, 0, stream>>>(
        obuf, bias, N_NODES * hc, hc);

    hin = obuf;
  }
}
